// Net_en_trans_65609920413745
// MI455X (gfx1250) — matrix operand order repaired and checked by reading, not compiled, not run on hardware
//
#include <hip/hip_runtime.h>
#include <math.h>

// ---------------------------------------------------------------------------
// Types for CDNA5 WMMA (wave32)
// ---------------------------------------------------------------------------
typedef _Float16 v16h __attribute__((ext_vector_type(16)));
typedef float    v8f  __attribute__((ext_vector_type(8)));

// ---------------------------------------------------------------------------
// Fragment helpers (CDNA5 16-bit WMMA VGPR layouts)
// ---------------------------------------------------------------------------
__device__ __forceinline__ void load_afrag(const float* __restrict__ Arow,
                                           int k0, int a_kgrp, v16h& af)
{
    #pragma unroll
    for (int r = 0; r < 8; ++r) {
        int kk = k0 + a_kgrp + 2 * r;
        float2 x = *(const float2*)(Arow + kk);      // global_load_b64
        af[2 * r]     = (_Float16)x.x;
        af[2 * r + 1] = (_Float16)x.y;
    }
}

// B loads via 32-bit element indices off the wave-uniform base W:
// backend emits global_load_b32 v, voffset, s[base] (scale_offset) instead of
// maintaining per-row 64-bit pointers.
__device__ __forceinline__ void load_bfrags(const float* __restrict__ W, int Nn,
                                            int k0, int b_kgrp, int b_col,
                                            v16h& bf0, v16h& bf1)
{
    #pragma unroll
    for (int r = 0; r < 8; ++r) {
        int idx = (k0 + b_kgrp + 2 * r) * Nn + b_col;
        bf0[2 * r]     = (_Float16)W[idx];
        bf0[2 * r + 1] = (_Float16)W[idx + Nn];
        bf1[2 * r]     = (_Float16)W[idx + 16];
        bf1[2 * r + 1] = (_Float16)W[idx + Nn + 16];
    }
}

__device__ __forceinline__ void store_tile(float* __restrict__ C,
                                           const float* __restrict__ bias, int Nn,
                                           int mbase, int n0,
                                           const v8f& c0, const v8f& c1)
{
    const float bv0 = bias[n0];
    const float bv1 = bias[n0 + 16];
    #pragma unroll
    for (int r = 0; r < 8; ++r) {
        const long mrow = (long)(mbase + r) * Nn;
        C[mrow + n0]      = c0[r] + bv0;
        C[mrow + n0 + 16] = c1[r] + bv1;
    }
}

// ---------------------------------------------------------------------------
// DUAL projection kernel: C1 = A@W1 + b1, C2 = A@W2 + b2 in one pass.
// A fragment loaded ONCE per K-step; 4 distinct B fragments are loaded first,
// then 4 independent v_wmma issue back-to-back (no WAR NOP spacing between).
// Requires M%16==0, K%32==0, Nn%32==0.
// ---------------------------------------------------------------------------
__global__ void __launch_bounds__(128, 1)
wmma_dual_kernel(const float* __restrict__ A, int M, int K, int Nn,
                 const float* __restrict__ W1, const float* __restrict__ b1,
                 float* __restrict__ C1,
                 const float* __restrict__ W2, const float* __restrict__ b2,
                 float* __restrict__ C2)
{
    const int lane = threadIdx.x & 31;
    const int wid  = threadIdx.x >> 5;
    const int tm   = (blockIdx.x * 4 + wid) * 16;
    const int tn   = blockIdx.y * 32;
    if (tm >= M) return;                       // wave-uniform exit

    v8f a10 = {}, a11 = {}, a20 = {}, a21 = {};

    const int am_row = tm + (lane & 15);
    const int a_kgrp = (lane & 16) ? 16 : 0;
    const int b_col  = tn + (lane & 15);
    const int b_kgrp = (lane & 16) ? 16 : 0;

    const float* __restrict__ Arow = A + (long)am_row * K;

    for (int k0 = 0; k0 < K; k0 += 32) {
        __builtin_prefetch(Arow + k0 + 32, 0, 1);   // speculative, safe

        v16h af;
        load_afrag(Arow, k0, a_kgrp, af);

        v16h b10, b11, b20, b21;                    // 4 distinct fragments
        load_bfrags(W1, Nn, k0, b_kgrp, b_col, b10, b11);
        load_bfrags(W2, Nn, k0, b_kgrp, b_col, b20, b21);

        a10 = __builtin_amdgcn_wmma_f32_16x16x32_f16(false, af, false, b10, (short)0, a10, false, false);
        a11 = __builtin_amdgcn_wmma_f32_16x16x32_f16(false, af, false, b11, (short)0, a11, false, false);
        a20 = __builtin_amdgcn_wmma_f32_16x16x32_f16(false, af, false, b20, (short)0, a20, false, false);
        a21 = __builtin_amdgcn_wmma_f32_16x16x32_f16(false, af, false, b21, (short)0, a21, false, false);
    }

    const int n0    = tn + (lane & 15);
    const int mbase = tm + ((lane & 16) ? 8 : 0);
    store_tile(C1, b1, Nn, mbase, n0, a10, a11);
    store_tile(C2, b2, Nn, mbase, n0, a20, a21);
}

// ---------------------------------------------------------------------------
// FAST WMMA GEMM (MLP head): C = epilogue(A @ W + bias); M%16, K%32, Nn%32.
// ---------------------------------------------------------------------------
__global__ void __launch_bounds__(128, 1)
wmma_gemm_fast_kernel(const float* __restrict__ A,
                      const float* __restrict__ W,
                      const float* __restrict__ bias,
                      const float* __restrict__ addm,
                      float* __restrict__ C,
                      int M, int K, int Nn, int relu)
{
    const int lane = threadIdx.x & 31;
    const int wid  = threadIdx.x >> 5;
    const int tm   = (blockIdx.x * 4 + wid) * 16;
    const int tn   = blockIdx.y * 32;
    if (tm >= M) return;

    v8f acc0 = {}, acc1 = {};
    const int am_row = tm + (lane & 15);
    const int a_kgrp = (lane & 16) ? 16 : 0;
    const int b_col  = tn + (lane & 15);
    const int b_kgrp = (lane & 16) ? 16 : 0;
    const float* __restrict__ Arow = A + (long)am_row * K;

    for (int k0 = 0; k0 < K; k0 += 32) {
        v16h af, bf0, bf1;
        load_afrag(Arow, k0, a_kgrp, af);
        load_bfrags(W, Nn, k0, b_kgrp, b_col, bf0, bf1);
        acc0 = __builtin_amdgcn_wmma_f32_16x16x32_f16(false, af, false, bf0, (short)0, acc0, false, false);
        acc1 = __builtin_amdgcn_wmma_f32_16x16x32_f16(false, af, false, bf1, (short)0, acc1, false, false);
    }

    const int   n0    = tn + (lane & 15);
    const int   mbase = tm + ((lane & 16) ? 8 : 0);
    const float bv0   = bias ? bias[n0]      : 0.0f;
    const float bv1   = bias ? bias[n0 + 16] : 0.0f;
    #pragma unroll
    for (int r = 0; r < 8; ++r) {
        const long mrow = (long)(mbase + r) * Nn;
        float v0 = acc0[r] + bv0;
        float v1 = acc1[r] + bv1;
        if (relu) { v0 = fmaxf(v0, 0.0f); v1 = fmaxf(v1, 0.0f); }
        if (addm) { v0 += addm[mrow + n0]; v1 += addm[mrow + n0 + 16]; }
        C[mrow + n0]      = v0;
        C[mrow + n0 + 16] = v1;
    }
}

// ---------------------------------------------------------------------------
// SAFE WMMA GEMM: arbitrary K / Nn (K=513, Nn=2 shapes; M%16==0).
// Branchless OOB: clamp index + 0/1 float mask (no per-load exec branches).
// ---------------------------------------------------------------------------
__global__ void __launch_bounds__(128, 1)
wmma_gemm_safe_kernel(const float* __restrict__ A,
                      const float* __restrict__ W,
                      const float* __restrict__ bias,
                      const float* __restrict__ addm,
                      float* __restrict__ C,
                      int M, int K, int Nn, int relu)
{
    const int lane = threadIdx.x & 31;
    const int wid  = threadIdx.x >> 5;
    const int tm   = (blockIdx.x * 4 + wid) * 16;
    const int tn   = blockIdx.y * 16;
    if (tm >= M) return;

    v8f acc = {};
    const int   am_row = tm + (lane & 15);
    const int   a_kgrp = (lane & 16) ? 16 : 0;
    const int   nraw   = tn + (lane & 15);
    const int   b_col  = (nraw < Nn) ? nraw : (Nn - 1);
    const float bmask  = (nraw < Nn) ? 1.0f : 0.0f;
    const int   b_kgrp = (lane & 16) ? 16 : 0;
    const float* __restrict__ Arow = A + (long)am_row * K;

    for (int k0 = 0; k0 < K; k0 += 32) {
        v16h af, bfrag;
        #pragma unroll
        for (int r = 0; r < 8; ++r) {
            int   kk  = k0 + a_kgrp + 2 * r;
            int   kc0 = (kk     < K) ? kk     : (K - 1);
            int   kc1 = (kk + 1 < K) ? kk + 1 : (K - 1);
            float m0  = (kk     < K) ? 1.0f : 0.0f;
            float m1  = (kk + 1 < K) ? 1.0f : 0.0f;
            af[2 * r]     = (_Float16)(Arow[kc0] * m0);
            af[2 * r + 1] = (_Float16)(Arow[kc1] * m1);
        }
        #pragma unroll
        for (int r = 0; r < 8; ++r) {
            int   kk  = k0 + b_kgrp + 2 * r;
            int   kc0 = (kk     < K) ? kk     : (K - 1);
            int   kc1 = (kk + 1 < K) ? kk + 1 : (K - 1);
            float m0  = ((kk     < K) ? 1.0f : 0.0f) * bmask;
            float m1  = ((kk + 1 < K) ? 1.0f : 0.0f) * bmask;
            bfrag[2 * r]     = (_Float16)(W[kc0 * Nn + b_col] * m0);
            bfrag[2 * r + 1] = (_Float16)(W[kc1 * Nn + b_col] * m1);
        }
        acc = __builtin_amdgcn_wmma_f32_16x16x32_f16(false, af, false, bfrag,
                                                     (short)0, acc, false, false);
    }

    if (nraw >= Nn) return;
    const int   mbase = tm + ((lane & 16) ? 8 : 0);
    const float bv    = bias ? bias[b_col] : 0.0f;
    #pragma unroll
    for (int r = 0; r < 8; ++r) {
        const long mrow = (long)(mbase + r) * Nn;
        float v = acc[r] + bv;
        if (relu) v = fmaxf(v, 0.0f);
        if (addm) v += addm[mrow + nraw];
        C[mrow + nraw] = v;
    }
}

// ---------------------------------------------------------------------------
// Edge-phase kernels (segment softmax over incoming edges per dst node)
// ---------------------------------------------------------------------------
__device__ __forceinline__ void atomicMaxF(float* addr, float val)
{
    if (val >= 0.0f) atomicMax((int*)addr, __float_as_int(val));
    else             atomicMin((unsigned int*)addr, __float_as_uint(val));
}

__global__ void edge_logits_kernel(const float* __restrict__ Q,
                                   const float* __restrict__ Kf,
                                   const float* __restrict__ We,
                                   const float* __restrict__ ea,
                                   const int* __restrict__ src,
                                   const int* __restrict__ dst,
                                   float* __restrict__ logits,
                                   float* __restrict__ mx,
                                   int E, int D, float scale)
{
    int e    = (int)((blockIdx.x * (long)blockDim.x + threadIdx.x) >> 5);
    int lane = threadIdx.x & 31;
    if (e >= E) return;
    int   s = src[e], d = dst[e];
    float a = ea[e];
    float part = 0.0f;
    for (int j = lane; j < D; j += 32)
        part += Q[(long)d * D + j] * (Kf[(long)s * D + j] + a * We[j]);
    for (int off = 16; off; off >>= 1) part += __shfl_xor(part, off, 32);
    if (lane == 0) {
        float lg = part * scale;
        logits[e] = lg;
        atomicMaxF(&mx[d], lg);
    }
}

__global__ void edge_exp_kernel(const float* __restrict__ mx,
                                const int* __restrict__ dst,
                                float* __restrict__ logits,
                                float* __restrict__ den, int E)
{
    int e = blockIdx.x * blockDim.x + threadIdx.x;
    if (e >= E) return;
    int   d = dst[e];
    float m = mx[d];
    if (!isfinite(m)) m = 0.0f;                 // empty-segment convention
    float w = __expf(logits[e] - m);
    logits[e] = w;
    atomicAdd(&den[d], w);
}

__global__ void edge_scatter_kernel(const float* __restrict__ V,
                                    const float* __restrict__ We,
                                    const float* __restrict__ ea,
                                    const int* __restrict__ src,
                                    const int* __restrict__ dst,
                                    const float* __restrict__ wgt,
                                    const float* __restrict__ den,
                                    float* __restrict__ agg,
                                    int E, int D)
{
    int e    = (int)((blockIdx.x * (long)blockDim.x + threadIdx.x) >> 5);
    int lane = threadIdx.x & 31;
    if (e >= E) return;
    int   s = src[e], d = dst[e];
    float a = ea[e];
    float coef = wgt[e] / (den[d] + 1e-16f);
    for (int j = lane; j < D; j += 32)
        atomicAdd(&agg[(long)d * D + j], coef * (V[(long)s * D + j] + a * We[j]));
}

// ---------------------------------------------------------------------------
// Small utility kernels
// ---------------------------------------------------------------------------
__global__ void fill_kernel(float* __restrict__ p, float v, long n)
{
    long i = blockIdx.x * (long)blockDim.x + threadIdx.x;
    if (i < n) p[i] = v;
}

__global__ void relu_copy_kernel(const float* __restrict__ in,
                                 float* __restrict__ out, long n)
{
    long i = blockIdx.x * (long)blockDim.x + threadIdx.x;
    if (i < n) out[i] = fmaxf(in[i], 0.0f);
}

__global__ void pool_kernel(const float* __restrict__ H,
                            const int* __restrict__ batchs,
                            float* __restrict__ g, int N, int D)
{
    long i = blockIdx.x * (long)blockDim.x + threadIdx.x;
    if (i >= (long)N * D) return;
    int row = (int)(i / D), col = (int)(i % D);
    atomicAdd(&g[(long)batchs[row] * D + col], H[i]);
}

__global__ void concat_kernel(const float* __restrict__ y,
                              const float* __restrict__ energy,
                              float* __restrict__ z, int G, int FCn)
{
    int i = blockIdx.x * blockDim.x + threadIdx.x;
    int tot = G * (FCn + 1);
    if (i >= tot) return;
    int row = i / (FCn + 1), col = i % (FCn + 1);
    z[i] = (col < FCn) ? y[(long)row * FCn + col] : energy[row];
}

__global__ void logsoftmax2_kernel(const float* __restrict__ z,
                                   float* __restrict__ out, int G)
{
    int g = blockIdx.x * blockDim.x + threadIdx.x;
    if (g >= G) return;
    float a = z[2 * g], b = z[2 * g + 1];
    float m = fmaxf(a, b);
    float l = m + logf(expf(a - m) + expf(b - m));
    out[2 * g]     = a - l;
    out[2 * g + 1] = b - l;
}

// ---------------------------------------------------------------------------
// Orchestration
// ---------------------------------------------------------------------------
extern "C" void kernel_launch(void* const* d_in, const int* in_sizes, int n_in,
                              void* d_out, int out_size, void* d_ws, size_t ws_size,
                              hipStream_t stream)
{
    const int F_IN = 128, D = 256, FCn = 512, Cc = 2, NL = 3;
    const int N = in_sizes[0] / F_IN;
    const int E = in_sizes[1];
    const int G = in_sizes[2];

    const float* x         = (const float*)d_in[0];
    const float* edge_attr = (const float*)d_in[1];
    const float* energy    = (const float*)d_in[2];
    const float* c1_Wq = (const float*)d_in[3];
    const float* c1_bq = (const float*)d_in[4];
    const float* c1_Wk = (const float*)d_in[5];
    const float* c1_bk = (const float*)d_in[6];
    const float* c1_Wv = (const float*)d_in[7];
    const float* c1_bv = (const float*)d_in[8];
    const float* c1_We = (const float*)d_in[9];
    const float* c1_Ws = (const float*)d_in[10];
    const float* c1_bs = (const float*)d_in[11];
    const float* cs_Wq = (const float*)d_in[12];
    const float* cs_bq = (const float*)d_in[13];
    const float* cs_Wk = (const float*)d_in[14];
    const float* cs_bk = (const float*)d_in[15];
    const float* cs_Wv = (const float*)d_in[16];
    const float* cs_bv = (const float*)d_in[17];
    const float* cs_We = (const float*)d_in[18];
    const float* cs_Ws = (const float*)d_in[19];
    const float* cs_bs = (const float*)d_in[20];
    const float* lin0_W = (const float*)d_in[21];
    const float* lin0_b = (const float*)d_in[22];
    const float* lin1_W = (const float*)d_in[23];
    const float* lin1_b = (const float*)d_in[24];
    const float* lin2_W = (const float*)d_in[25];
    const float* lin2_b = (const float*)d_in[26];
    const float* lin3_W = (const float*)d_in[27];
    const float* lin3_b = (const float*)d_in[28];
    const int* edge_index = (const int*)d_in[29];
    const int* batchs     = (const int*)d_in[30];
    const int* srcp = edge_index;
    const int* dstp = edge_index + E;

    // -------- workspace carve (floats) --------
    float* ws  = (float*)d_ws;
    float* Q   = ws; ws += (long)N * D;
    float* Kb  = ws; ws += (long)N * D;
    float* Vb  = ws; ws += (long)N * D;
    float* H   = ws; ws += (long)N * D;
    float* AGG = ws; ws += (long)N * D;
    float* LG  = ws; ws += E;
    float* MX  = ws; ws += N;
    float* DEN = ws; ws += N;
    float* GP  = ws; ws += (long)G * D;
    float* Y   = ws; ws += (long)G * FCn;
    float* ZB  = ws; ws += (long)G * (FCn + 1);
    float* Z1  = ws; ws += (long)G * FCn;
    float* Z2  = ws; ws += (long)G * FCn;
    float* Z3  = ws; ws += (long)G * Cc;

    auto gemm_dual = [&](const float* A, int M, int K, int Nn,
                         const float* W1, const float* b1, float* C1,
                         const float* W2, const float* b2, float* C2) {
        dim3 grid((M + 63) / 64, Nn / 32);
        wmma_dual_kernel<<<grid, 128, 0, stream>>>(A, M, K, Nn, W1, b1, C1, W2, b2, C2);
    };
    auto gemm_fast = [&](const float* A, const float* W, const float* bias,
                         const float* addm, float* Cout, int M, int K, int Nn, int relu) {
        dim3 grid((M + 63) / 64, Nn / 32);
        wmma_gemm_fast_kernel<<<grid, 128, 0, stream>>>(A, W, bias, addm, Cout,
                                                        M, K, Nn, relu);
    };
    auto gemm_safe = [&](const float* A, const float* W, const float* bias,
                         const float* addm, float* Cout, int M, int K, int Nn, int relu) {
        dim3 grid((M + 63) / 64, (Nn + 15) / 16);
        wmma_gemm_safe_kernel<<<grid, 128, 0, stream>>>(A, W, bias, addm, Cout,
                                                        M, K, Nn, relu);
    };
    auto fill = [&](float* p, float v, long n) {
        fill_kernel<<<(unsigned)((n + 255) / 256), 256, 0, stream>>>(p, v, n);
    };

    const long ND      = (long)N * D;
    const int  eWaveBl = (int)(((long)E * 32 + 255) / 256);   // wave-per-edge grids
    const float scale  = 0.0625f;                             // 1/sqrt(D), D=256

    // -------- TransformerConv stack --------
    const float* in = x;
    int Kdim = F_IN;
    for (int layer = 0; layer <= NL; ++layer) {
        const float *Wq, *bq, *Wk, *bk, *Wv, *bv, *We, *Wsk, *bsk;
        if (layer == 0) {
            Wq = c1_Wq; bq = c1_bq; Wk = c1_Wk; bk = c1_bk;
            Wv = c1_Wv; bv = c1_bv; We = c1_We; Wsk = c1_Ws; bsk = c1_bs;
        } else {
            int i = layer - 1;
            Wq  = cs_Wq + (long)i * D * D;  bq  = cs_bq + (long)i * D;
            Wk  = cs_Wk + (long)i * D * D;  bk  = cs_bk + (long)i * D;
            Wv  = cs_Wv + (long)i * D * D;  bv  = cs_bv + (long)i * D;
            We  = cs_We + (long)i * D;
            Wsk = cs_Ws + (long)i * D * D;  bsk = cs_bs + (long)i * D;
        }
        // Fused projections (2+2): Q/K in one pass over A, V/skip in another.
        gemm_dual(in, N, Kdim, D, Wq, bq, Q,  Wk,  bk,  Kb);
        gemm_dual(in, N, Kdim, D, Wv, bv, Vb, Wsk, bsk, AGG);

        // Segment softmax (3 passes) + scatter into AGG.
        fill(MX, -INFINITY, N);
        fill(DEN, 0.0f, N);
        edge_logits_kernel<<<eWaveBl, 256, 0, stream>>>(Q, Kb, We, edge_attr,
                                                        srcp, dstp, LG, MX, E, D, scale);
        edge_exp_kernel<<<(E + 255) / 256, 256, 0, stream>>>(MX, dstp, LG, DEN, E);
        edge_scatter_kernel<<<eWaveBl, 256, 0, stream>>>(Vb, We, edge_attr, srcp, dstp,
                                                         LG, DEN, AGG, E, D);
        // h = relu(attn + skip)
        relu_copy_kernel<<<(unsigned)((ND + 255) / 256), 256, 0, stream>>>(AGG, H, ND);
        in = H; Kdim = D;
    }

    // -------- global add pool --------
    fill(GP, 0.0f, (long)G * D);
    pool_kernel<<<(unsigned)((ND + 255) / 256), 256, 0, stream>>>(H, batchs, GP, N, D);

    // -------- MLP head --------
    gemm_fast(GP, lin0_W, lin0_b, nullptr, Y, G, D, FCn, 1);            // y = relu(g@W0+b0)
    concat_kernel<<<(G * (FCn + 1) + 255) / 256, 256, 0, stream>>>(Y, energy, ZB, G, FCn);
    gemm_safe(ZB, lin1_W, lin1_b, Y,       Z1, G, FCn + 1, FCn, 1);     // relu(z@W1+b1)+y
    gemm_fast(Z1, lin2_W, lin2_b, nullptr, Z2, G, FCn,     FCn, 1);     // relu(..)
    gemm_safe(Z2, lin3_W, lin3_b, nullptr, Z3, G, FCn,     Cc,  0);     // logits
    logsoftmax2_kernel<<<(G + 255) / 256, 256, 0, stream>>>(Z3, (float*)d_out, G);
}
